// LSTM_10007273800410
// MI455X (gfx1250) — matrix operand order repaired and checked by reading, not compiled, not run on hardware
//
#include <hip/hip_runtime.h>

// Problem constants (match reference): T=512, B=64, E=1024, H=1024
#define TT 512
#define BB 64
#define EE 1024
#define HH 1024
// K = E+H = 2048 -> 64 k-tiles of 32 ;  N = 4H = 4096 -> 256 n-tiles of 16

typedef __attribute__((ext_vector_type(8)))  float          v8f;
typedef __attribute__((ext_vector_type(16))) __bf16         v16bf;
typedef __attribute__((ext_vector_type(16))) unsigned short v16us;
typedef __attribute__((ext_vector_type(8)))  unsigned short v8us;

__device__ __forceinline__ unsigned short f2bf(float f) {
  unsigned u = __builtin_bit_cast(unsigned, f);
  u += 0x7FFFu + ((u >> 16) & 1u);          // round-to-nearest-even
  return (unsigned short)(u >> 16);
}
__device__ __forceinline__ float fast_sigmoid(float x) {
  return 1.0f / (1.0f + __expf(-x));
}
__device__ __forceinline__ float fast_tanh(float x) {
  return 2.0f / (1.0f + __expf(-2.0f * x)) - 1.0f;
}

// Async copy 16 bytes / lane, global -> LDS (tracked with ASYNCcnt).
__device__ __forceinline__ void async_cp16(unsigned lds_addr, const void* gaddr) {
  asm volatile("global_load_async_to_lds_b128 %0, %1, off"
               :: "v"(lds_addr), "v"(gaddr) : "memory");
}

// ---------------------------------------------------------------------------
// One-time: pack W = [Wi;Wf;Wg;Wo] (fp32, each [H, E+H]) into bf16
// B-fragment layout (contiguous K, matching A loads) for v_wmma_f32_16x16x32_bf16:
//   frag(ntile, ktile): 32 lanes x 16 bf16, lane&15 = N-col, lanes>=16 hold
//   K_local 16..31, elems are consecutive K. Linear index:
//   ((ntile*64 + ktile)*32 + lane)*16 + elem
// ---------------------------------------------------------------------------
__global__ void pack_w_kernel(const float* __restrict__ Wi,
                              const float* __restrict__ Wf,
                              const float* __restrict__ Wg,
                              const float* __restrict__ Wo,
                              unsigned short* __restrict__ Wp) {
  unsigned gid   = blockIdx.x * blockDim.x + threadIdx.x;   // [0, 256*64*32)
  unsigned lane  = gid & 31;
  unsigned kt    = (gid >> 5) & 63;
  unsigned ntile = gid >> 11;                               // [0, 256)
  unsigned n     = ntile * 16 + (lane & 15);                // output column
  unsigned gate  = n >> 10;
  unsigned row   = n & 1023;
  const float* W = (gate == 0) ? Wi : (gate == 1) ? Wf : (gate == 2) ? Wg : Wo;
  const float* src = W + (size_t)row * (EE + HH) + kt * 32 + ((lane >> 4) << 4);
  v16us out;
#pragma unroll
  for (int e = 0; e < 16; ++e) out[e] = f2bf(src[e]);
  *(v16us*)(Wp + (size_t)gid * 16) = out;
}

// init: h0 = bf16(hx), c = cx, xbf0 = bf16(emb[0])
__global__ void init_state_kernel(const float* __restrict__ hx,
                                  const float* __restrict__ cx,
                                  const float* __restrict__ x0,
                                  unsigned short* __restrict__ h0,
                                  float* __restrict__ c0,
                                  unsigned short* __restrict__ xbf0) {
  int i = blockIdx.x * blockDim.x + threadIdx.x;
  h0[i]   = f2bf(hx[i]);
  c0[i]   = cx[i];
  xbf0[i] = f2bf(x0[i]);
}

// ---------------------------------------------------------------------------
// Fused LSTM step: bf16 WMMA GEMM (f32 accum) + gate math + cell update.
// Wave w = (jt, mpair): batch rows [mpair*32, +32), hidden cols [jt*16, +16),
// all 4 gates (8 C tiles, 512 WMMAs/wave). B fragments are staged through
// LDS via global_load_async_to_lds_b128, double-buffered per wave.
// Epilogue also converts next step's x slice to bf16 (each wave owns a
// disjoint 32x16 slice of [64,1024]).
// ---------------------------------------------------------------------------
__global__ __launch_bounds__(128)
void lstm_step_kernel(const unsigned short* __restrict__ xbf_t, // bf16 [B,E]
                      const unsigned short* __restrict__ Wp,
                      const float* __restrict__ bi, const float* __restrict__ bf_,
                      const float* __restrict__ bg, const float* __restrict__ bo,
                      const unsigned short* __restrict__ h_in,  // bf16 [B,H]
                      unsigned short* __restrict__ h_out,       // bf16 [B,H]
                      float* __restrict__ c_state,              // f32 [B,H]
                      float* __restrict__ out_t,                // out + t*B*H
                      const float* __restrict__ x_next,         // emb[t+1] or null
                      unsigned short* __restrict__ xbf_next,    // bf16 [B,E]
                      float* __restrict__ outh,                 // last step only
                      float* __restrict__ outc) {
  // [wave][buf][gate][512 bf16] = 32 KB / block
  __shared__ unsigned short sh[4][2][4][512];

  const int lane  = threadIdx.x & 31;
  const int wl    = threadIdx.x >> 5;      // wave in block
  const int w     = blockIdx.x * (blockDim.x >> 5) + wl;
  const int jt    = w >> 1;                // [0,64)
  const int mpair = w & 1;                 // [0,2)
  const int lhalf = lane >> 4;             // 0/1
  const int lmod  = lane & 15;

  // Per-gate global base for this wave's B column-tile (advance by kt*512).
  const unsigned short* wbase[4];
#pragma unroll
  for (int g = 0; g < 4; ++g)
    wbase[g] = Wp + (((size_t)(g * 64 + jt)) * 64) * 512 + lane * 16;

  // Issue one k-tile's B fragments (4 gates x 32B/lane = 8 async b128).
  auto issueB = [&](int kt) {
    const int buf = kt & 1;
#pragma unroll
    for (int g = 0; g < 4; ++g) {
      unsigned lds = (unsigned)(size_t)(&sh[wl][buf][g][0]) + (unsigned)lane * 32u;
      const unsigned short* gp = wbase[g] + (size_t)kt * 512;
      async_cp16(lds, gp);
      async_cp16(lds + 16u, gp + 8);
    }
  };

  v8f acc[4][2];
#pragma unroll
  for (int g = 0; g < 4; ++g)
#pragma unroll
    for (int mt = 0; mt < 2; ++mt) acc[g][mt] = {};

  issueB(0);

  for (int kt = 0; kt < 64; ++kt) {
    // A frags straight from bf16 memory (contiguous K order as in packed B: lanes0-15 take
    // K k0..k0+15; lanes16-31 shift base by +16).
    const int kb = (kt & 31) * 32 + (lhalf ? 16 : 0);
    const unsigned short* srcbase = (kt < 32) ? xbf_t : h_in;
    v16bf afrag[2];
#pragma unroll
    for (int mt = 0; mt < 2; ++mt) {
      const unsigned short* s =
          srcbase + (size_t)(mpair * 32 + mt * 16 + lmod) * 1024 + kb;
      v8us lo = *(const v8us*)(s);        // elems 0..7  (K = kb .. kb+7)
      v8us hi = *(const v8us*)(s + 8);    // elems 8..15 (K = kb+8 .. kb+15)
      v16us u = __builtin_shufflevector(lo, hi, 0, 1, 2, 3, 4, 5, 6, 7,
                                        8, 9, 10, 11, 12, 13, 14, 15);
      afrag[mt] = __builtin_bit_cast(v16bf, u);
    }

    // Pipeline: launch next tile's copies, then wait for current tile.
    if (kt < 63) {
      issueB(kt + 1);
      asm volatile("s_wait_asynccnt 8" ::: "memory");
    } else {
      asm volatile("s_wait_asynccnt 0" ::: "memory");
    }

    // Consume B from LDS (ds_load_b128 x2 per gate); 8 WMMAs per k-tile.
    const int buf = kt & 1;
#pragma unroll
    for (int g = 0; g < 4; ++g) {
      v16bf bfrag = __builtin_bit_cast(
          v16bf, *(const v16us*)(&sh[wl][buf][g][lane * 16]));
#pragma unroll
      for (int mt = 0; mt < 2; ++mt)
        acc[g][mt] = __builtin_amdgcn_wmma_f32_16x16x32_bf16(
            false, afrag[mt], false, bfrag, (short)0, acc[g][mt], false, false);
    }
  }

  // Cell update. C layout: VGPR r -> M = r (lanes<16) / 8+r; lane&15 -> N.
  const int   j   = jt * 16 + lmod;
  const float vbi = bi[j], vbf = bf_[j], vbg = bg[j], vbo = bo[j];
#pragma unroll
  for (int mt = 0; mt < 2; ++mt) {
#pragma unroll
    for (int r = 0; r < 8; ++r) {
      const int    row = mpair * 32 + mt * 16 + lhalf * 8 + r;
      const size_t idx = (size_t)row * HH + j;
      float iv = fast_sigmoid(acc[0][mt][r] + vbi);
      float fv = fast_sigmoid(acc[1][mt][r] + vbf);
      float gv = fast_tanh   (acc[2][mt][r] + vbg);
      float ov = fast_sigmoid(acc[3][mt][r] + vbo);
      float c2 = fv * c_state[idx] + iv * gv;
      float h2 = ov * fast_tanh(c2);
      c_state[idx] = c2;
      h_out[idx]   = f2bf(h2);
      out_t[idx]   = h2;
      if (x_next) xbf_next[idx] = f2bf(x_next[idx]);  // stage next x (E==H)
      if (outh) { outh[idx] = h2; outc[idx] = c2; }
    }
  }
}

// ---------------------------------------------------------------------------
extern "C" void kernel_launch(void* const* d_in, const int* in_sizes, int n_in,
                              void* d_out, int out_size, void* d_ws, size_t ws_size,
                              hipStream_t stream) {
  const float* emb = (const float*)d_in[0];
  const float* hx  = (const float*)d_in[1];
  const float* cx  = (const float*)d_in[2];
  const float* Wi  = (const float*)d_in[3];
  const float* Wf  = (const float*)d_in[4];
  const float* Wg  = (const float*)d_in[5];
  const float* Wo  = (const float*)d_in[6];
  const float* bi  = (const float*)d_in[7];
  const float* bf  = (const float*)d_in[8];
  const float* bg  = (const float*)d_in[9];
  const float* bo  = (const float*)d_in[10];
  float* out = (float*)d_out;

  char* ws = (char*)d_ws;
  const size_t WP_BYTES = (size_t)256 * 64 * 512 * sizeof(unsigned short); // 16 MB
  const size_t HB_BYTES = (size_t)BB * HH * sizeof(unsigned short);        // 128 KB
  unsigned short* Wp    = (unsigned short*)ws;
  unsigned short* hbuf0 = (unsigned short*)(ws + WP_BYTES);
  unsigned short* hbuf1 = (unsigned short*)(ws + WP_BYTES + HB_BYTES);
  unsigned short* xbuf0 = (unsigned short*)(ws + WP_BYTES + 2 * HB_BYTES);
  unsigned short* xbuf1 = (unsigned short*)(ws + WP_BYTES + 3 * HB_BYTES);
  float*          cbuf  = (float*)(ws + WP_BYTES + 4 * HB_BYTES);

  // One-time per launch: pack weights, init state + first x slice.
  pack_w_kernel<<<2048, 256, 0, stream>>>(Wi, Wf, Wg, Wo, Wp);
  init_state_kernel<<<(BB * HH) / 256, 256, 0, stream>>>(hx, cx, emb,
                                                         hbuf0, cbuf, xbuf0);

  for (int t = 0; t < TT; ++t) {
    const unsigned short* xbf_in = (t & 1) ? xbuf1 : xbuf0;
    unsigned short* xbf_nx       = (t & 1) ? xbuf0 : xbuf1;
    const unsigned short* h_in   = (t & 1) ? hbuf1 : hbuf0;
    unsigned short* h_out        = (t & 1) ? hbuf0 : hbuf1;
    const float* x_next = (t < TT - 1) ? emb + (size_t)(t + 1) * BB * EE : nullptr;
    float* out_t = out + (size_t)t * BB * HH;
    float* outh  = (t == TT - 1) ? out + (size_t)TT * BB * HH : nullptr;
    float* outc  = (t == TT - 1) ? out + (size_t)TT * BB * HH + BB * HH : nullptr;
    lstm_step_kernel<<<32, 128, 0, stream>>>(xbf_in, Wp, bi, bf, bg, bo,
                                             h_in, h_out, cbuf, out_t,
                                             x_next, xbf_nx, outh, outc);
  }
}
